// GAT_51788715655952
// MI455X (gfx1250) — compile-verified, matrix operand order checked by reading, not run on hardware
//
#include <hip/hip_runtime.h>
#include <hip/hip_bf16.h>

// ---------------------------------------------------------------------------
// GAT (2 × MPGATConv blocks + mean-pool + linear head) for gfx1250 (MI455X).
// All matmuls (node projection, per-edge edge_lin, block tail) run on
// v_wmma_f32_16x16x32_f16 (f16 A/B, f32 accumulate). Compile-time K sizes so
// fragment loads are straight-line float4 (global_load_b128) with no
// exec-divergent guards. Scatter/softmax via global f32/u32 atomics.
// ---------------------------------------------------------------------------

typedef __attribute__((ext_vector_type(16))) _Float16 v16h;
typedef __attribute__((ext_vector_type(8)))  float    v8f;

#define HID 32

__device__ __forceinline__ v8f wmma_f16(v16h a, v16h b, v8f c) {
    // (neg_a, A, neg_b, B, c_mod, C, reuse_a, reuse_b)
    return __builtin_amdgcn_wmma_f32_16x16x32_f16(false, a, false, b, (short)0, c,
                                                  false, false);
}

// A/B fragment element t of lane l covers K = (t&7) + 16*(t>>3) + 8*(l>>4)
__device__ __forceinline__ int kmap(int t, int half) {
    return (t & 7) + ((t >> 3) << 4) + (half << 3);
}

__device__ __forceinline__ float4 ld4(const float* p) {
    return *(const float4*)p;
}

// Pack 8 consecutive floats (two float4) * m into fragment elements t0..t0+7.
__device__ __forceinline__ void pack8(v16h& a, int t0, const float* p, float m) {
    float4 q0 = ld4(p), q1 = ld4(p + 4);
    a[t0 + 0] = (_Float16)(q0.x * m); a[t0 + 1] = (_Float16)(q0.y * m);
    a[t0 + 2] = (_Float16)(q0.z * m); a[t0 + 3] = (_Float16)(q0.w * m);
    a[t0 + 4] = (_Float16)(q1.x * m); a[t0 + 5] = (_Float16)(q1.y * m);
    a[t0 + 6] = (_Float16)(q1.z * m); a[t0 + 7] = (_Float16)(q1.w * m);
}

// A fragment from a row of length D (compile-time), K-step base k0.
// Runs are 8-aligned; D is a multiple of 8 for all call sites, so each 8-run
// is all-in or all-out. Out-of-range runs load from a safe address and are
// zeroed via the multiplier (no exec-divergent loads).
template <int D>
__device__ __forceinline__ v16h load_afrag(const float* __restrict__ rowp,
                                           int k0, int half, float scale) {
    const int r0 = k0 + half * 8;
    const int r1 = k0 + 16 + half * 8;
    const bool ok0 = (r0 + 8) <= D;
    const bool ok1 = (r1 + 8) <= D;
    v16h a;
    pack8(a, 0, ok0 ? rowp + r0 : rowp, ok0 ? scale : 0.f);
    pack8(a, 8, ok1 ? rowp + r1 : rowp, ok1 ? scale : 0.f);
    return a;
}

// A fragment from a full 32-float row (no guards).
__device__ __forceinline__ v16h load_afrag32(const float* __restrict__ rowp,
                                             int half, float scale) {
    v16h a;
    pack8(a, 0, rowp + half * 8,      scale);
    pack8(a, 8, rowp + 16 + half * 8, scale);
    return a;
}

// B fragment: row-major W[D x 32], columns n0..n0+15, K-step base k0.
// Clamp-index + value-select instead of guarded loads.
template <int D>
__device__ __forceinline__ v16h load_bfrag(const float* __restrict__ W,
                                           int n0, int k0, int lane) {
    const int col  = n0 + (lane & 15);
    const int half = lane >> 4;
    v16h b;
#pragma unroll
    for (int t = 0; t < 16; ++t) {
        const int k  = k0 + kmap(t, half);
        const int kk = (k < D) ? k : 0;
        const float v = W[kk * HID + col];
        b[t] = (k < D) ? (_Float16)v : (_Float16)0.f;
    }
    return b;
}

// ---------------------------------------------------------------------------
// xp[N,32] = X[N,D] @ W[D,32]          (one 16x32 output tile per wave)
// ---------------------------------------------------------------------------
template <int D>
__global__ void proj_wmma(const float* __restrict__ X, const float* __restrict__ W,
                          float* __restrict__ XP, int N) {
    const int wv   = threadIdx.x >> 5;
    const int lane = threadIdx.x & 31;
    const int tile = blockIdx.x * (blockDim.x >> 5) + wv;
    const int row0 = tile * 16;
    if (row0 >= N) return;                       // wave-uniform
    const int half = lane >> 4, r = lane & 15;
    const float* rowp = X + (size_t)(row0 + r) * D;

    v8f acc0 = {}, acc1 = {};
    constexpr int KSTEPS = (D + 31) / 32;
#pragma unroll
    for (int s = 0; s < KSTEPS; ++s) {
        const int k0 = s * 32;
        v16h a  = load_afrag<D>(rowp, k0, half, 1.f);
        v16h b0 = load_bfrag<D>(W, 0,  k0, lane);
        v16h b1 = load_bfrag<D>(W, 16, k0, lane);
        acc0 = wmma_f16(a, b0, acc0);
        acc1 = wmma_f16(a, b1, acc1);
    }
#pragma unroll
    for (int v = 0; v < 8; ++v) {
        const size_t row = (size_t)(row0 + v + 8 * half);
        XP[row * HID + r]      = acc0[v];
        XP[row * HID + 16 + r] = acc1[v];
    }
}

// ---------------------------------------------------------------------------
// per-node attention dots
// ---------------------------------------------------------------------------
__global__ void alpha_kernel(const float* __restrict__ XP,
                             const float* __restrict__ asrc,
                             const float* __restrict__ adst,
                             float* __restrict__ als, float* __restrict__ ald,
                             int N) {
    int n = blockIdx.x * blockDim.x + threadIdx.x;
    if (n >= N) return;
    const float* row = XP + (size_t)n * HID;
    float s = 0.f, d = 0.f;
#pragma unroll
    for (int q = 0; q < HID / 4; ++q) {
        float4 v = ld4(row + q * 4);
        float4 as = ld4(asrc + q * 4);
        float4 ad = ld4(adst + q * 4);
        s += v.x * as.x + v.y * as.y + v.z * as.z + v.w * as.w;
        d += v.x * ad.x + v.y * ad.y + v.z * ad.z + v.w * ad.w;
    }
    als[n] = s; ald[n] = d;
}

// ---------------------------------------------------------------------------
// edge logits + segment-max via order-preserving uint atomicMax
// ---------------------------------------------------------------------------
__device__ __forceinline__ unsigned fenc(float f) {
    unsigned b = __float_as_uint(f);
    return (b & 0x80000000u) ? ~b : (b | 0x80000000u);
}

__global__ void edge_logits(const int* __restrict__ ei,
                            const float* __restrict__ als,
                            const float* __restrict__ ald,
                            float* __restrict__ a, unsigned* __restrict__ mEnc,
                            int E) {
    int e = blockIdx.x * blockDim.x + threadIdx.x;
    if (e >= E) return;
    int s = ei[e], d = ei[E + e];
    float v = als[s] + ald[d];
    v = (v > 0.f) ? v : 0.2f * v;
    a[e] = v;
    atomicMax(&mEnc[d], fenc(v));
}

__global__ void decode_m(const unsigned* __restrict__ mEnc,
                         float* __restrict__ mF, int N) {
    int n = blockIdx.x * blockDim.x + threadIdx.x;
    if (n >= N) return;
    unsigned u = mEnc[n];
    unsigned bits = (u & 0x80000000u) ? (u & 0x7FFFFFFFu) : ~u;
    // non-finite (no incoming edge) -> 0, matching the reference
    mF[n] = ((bits & 0x7F800000u) == 0x7F800000u) ? 0.f : __uint_as_float(bits);
}

__global__ void edge_expsum(const int* __restrict__ ei,
                            const float* __restrict__ a,
                            const float* __restrict__ mF,
                            float* __restrict__ sSum, int E) {
    int e = blockIdx.x * blockDim.x + threadIdx.x;
    if (e >= E) return;
    int d = ei[E + e];
    atomicAdd(&sSum[d], __expf(a[e] - mF[d]));
}

// ---------------------------------------------------------------------------
// per-edge message GEMM: [16 edges x 65] @ elw[65 x 32], f16 WMMA, K-steps:
//   0: xp[dst][0..31]   1: xp[src][0..31]*alpha   2: |ew| (K=64), zero-padded
// B fragments loaded once per wave, reused over MSG_TILES*16 edges.
// D tile scattered into agg with f32 atomics (+elb per edge).
// ---------------------------------------------------------------------------
#define MSG_TILES 16
__global__ void msg_wmma(const int* __restrict__ ei,
                         const float* __restrict__ edge_attr,
                         const float* __restrict__ a,
                         const float* __restrict__ mF,
                         const float* __restrict__ sSum,
                         const float* __restrict__ XP,
                         const float* __restrict__ elw,
                         const float* __restrict__ elb,
                         float* __restrict__ agg, int E) {
    const int wv   = threadIdx.x >> 5;
    const int lane = threadIdx.x & 31;
    const int half = lane >> 4, r = lane & 15;
    const long tile0 = ((long)blockIdx.x * (blockDim.x >> 5) + wv) * MSG_TILES;

    // B fragments (elw, 65x32) live in VGPRs for all 3 K-steps x 2 N-tiles.
    v16h B[3][2];
#pragma unroll
    for (int s = 0; s < 3; ++s) {
        B[s][0] = load_bfrag<2 * HID + 1>(elw, 0,  s * 32, lane);
        B[s][1] = load_bfrag<2 * HID + 1>(elw, 16, s * 32, lane);
    }
    const float elb0 = elb[r], elb1 = elb[16 + r];

#pragma unroll 1
    for (int t = 0; t < MSG_TILES; ++t) {
        const long e0 = (tile0 + t) * 16;
        if (e0 >= E) break;                      // wave-uniform
        __builtin_prefetch(&ei[E + e0 + 512], 0, 1);

        const int e   = (int)e0 + r;             // both lane-halves: same edge row
        const int src = ei[e], dst = ei[E + e];
        float denom = sSum[dst];
        denom = (denom == 0.f) ? 1.f : denom;
        const float alpha = __expf(a[e] - mF[dst]) / denom;
        const float ew    = fabsf(edge_attr[e]);
        const float* xd = XP + (size_t)dst * HID;
        const float* xs = XP + (size_t)src * HID;

        v8f acc0 = {}, acc1 = {};
        v16h af0 = load_afrag32(xd, half, 1.f);      // K 0..31  : x_i
        acc0 = wmma_f16(af0, B[0][0], acc0);
        acc1 = wmma_f16(af0, B[0][1], acc1);
        v16h af1 = load_afrag32(xs, half, alpha);    // K 32..63 : alpha * x_j
        acc0 = wmma_f16(af1, B[1][0], acc0);
        acc1 = wmma_f16(af1, B[1][1], acc1);
        v16h af2 = {};                               // K 64     : |edge_w|
        af2[0] = (half == 0) ? (_Float16)ew : (_Float16)0.f;
        acc0 = wmma_f16(af2, B[2][0], acc0);
        acc1 = wmma_f16(af2, B[2][1], acc1);

        // D row v belongs to edge e0 + v + 8*half; scatter to agg[dst_of_row]
#pragma unroll
        for (int v = 0; v < 8; ++v) {
            const long erow = e0 + v + 8 * half;
            const int  drow = ei[E + erow];
            atomicAdd(&agg[(size_t)drow * HID + r],      acc0[v] + elb0);
            atomicAdd(&agg[(size_t)drow * HID + 16 + r], acc1[v] + elb1);
        }
    }
}

// ---------------------------------------------------------------------------
// block tail: z = BN(leaky_relu((agg + b) @ pw + pb))   (single K-step WMMA)
// ---------------------------------------------------------------------------
__global__ void tail_wmma(const float* __restrict__ agg,
                          const float* __restrict__ bbias,
                          const float* __restrict__ pw,
                          const float* __restrict__ pb,
                          const float* __restrict__ bng,
                          const float* __restrict__ bnb,
                          float* __restrict__ Z, int N) {
    const int wv   = threadIdx.x >> 5;
    const int lane = threadIdx.x & 31;
    const int tile = blockIdx.x * (blockDim.x >> 5) + wv;
    const int row0 = tile * 16;
    if (row0 >= N) return;
    const int half = lane >> 4, r = lane & 15;
    const float* rowp = agg + (size_t)(row0 + r) * HID;

    // A = agg_row + b   (two 8-float runs, both vectorized)
    v16h af;
#pragma unroll
    for (int run = 0; run < 2; ++run) {
        const int base = run * 16 + half * 8;
        float4 x0 = ld4(rowp + base),     x1 = ld4(rowp + base + 4);
        float4 b0 = ld4(bbias + base),    b1 = ld4(bbias + base + 4);
        const int t0 = run * 8;
        af[t0 + 0] = (_Float16)(x0.x + b0.x); af[t0 + 1] = (_Float16)(x0.y + b0.y);
        af[t0 + 2] = (_Float16)(x0.z + b0.z); af[t0 + 3] = (_Float16)(x0.w + b0.w);
        af[t0 + 4] = (_Float16)(x1.x + b1.x); af[t0 + 5] = (_Float16)(x1.y + b1.y);
        af[t0 + 6] = (_Float16)(x1.z + b1.z); af[t0 + 7] = (_Float16)(x1.w + b1.w);
    }
    v16h wb0 = load_bfrag<HID>(pw, 0,  0, lane);
    v16h wb1 = load_bfrag<HID>(pw, 16, 0, lane);
    v8f acc0 = {}, acc1 = {};
    acc0 = wmma_f16(af, wb0, acc0);
    acc1 = wmma_f16(af, wb1, acc1);

    const float invs = 0.9999950000374997f;       // 1/sqrt(1 + 1e-5)
    const float g0 = bng[r] * invs, g1 = bng[16 + r] * invs;
    const float p0 = pb[r],  p1 = pb[16 + r];
    const float c0 = bnb[r], c1 = bnb[16 + r];
#pragma unroll
    for (int v = 0; v < 8; ++v) {
        const size_t row = (size_t)(row0 + v + 8 * half);
        float z0 = acc0[v] + p0; z0 = (z0 > 0.f) ? z0 : 0.2f * z0;
        Z[row * HID + r] = z0 * g0 + c0;
        float z1 = acc1[v] + p1; z1 = (z1 > 0.f) ? z1 : 0.2f * z1;
        Z[row * HID + 16 + r] = z1 * g1 + c1;
    }
}

// ---------------------------------------------------------------------------
// housekeeping / pooling / head
// ---------------------------------------------------------------------------
__global__ void clear_block_state(unsigned* __restrict__ mEnc,
                                  float* __restrict__ sSum,
                                  float* __restrict__ agg, int N) {
    int i = blockIdx.x * blockDim.x + threadIdx.x;
    if (i < N) { mEnc[i] = 0x007FFFFFu; /* enc(-inf) */ sSum[i] = 0.f; }
    if (i < N * HID) agg[i] = 0.f;
}

__global__ void clear_pool(float* __restrict__ pool, int n) {
    int i = blockIdx.x * blockDim.x + threadIdx.x;
    if (i < n) pool[i] = 0.f;
}

__global__ void pool_kernel(const float* __restrict__ Z,
                            const int* __restrict__ batch,
                            float* __restrict__ tot, float* __restrict__ cnt,
                            int N) {
    int i = blockIdx.x * blockDim.x + threadIdx.x;
    if (i >= N * HID) return;
    int n = i >> 5, k = i & 31;
    int g = batch[n];
    atomicAdd(&tot[g * HID + k], Z[i]);
    if (k == 0) atomicAdd(&cnt[g], 1.0f);
}

__global__ void head_kernel(const float* __restrict__ tot,
                            const float* __restrict__ cnt,
                            const float* __restrict__ fw,
                            const float* __restrict__ fb,
                            float* __restrict__ out, int NG, int NC) {
    int t = blockIdx.x * blockDim.x + threadIdx.x;
    if (t >= NG * NC) return;
    int g = t / NC, c = t % NC;
    float cc = cnt[g]; cc = (cc > 1.f) ? cc : 1.f;
    float acc = fb[c];
#pragma unroll
    for (int k = 0; k < HID; ++k) acc += (tot[g * HID + k] / cc) * fw[k * NC + c];
    out[g * NC + c] = acc;
}

// ---------------------------------------------------------------------------
extern "C" void kernel_launch(void* const* d_in, const int* in_sizes, int n_in,
                              void* d_out, int out_size, void* d_ws, size_t ws_size,
                              hipStream_t stream) {
    const float* x   = (const float*)d_in[0];
    const float* ea  = (const float*)d_in[1];
    const int*   ei  = (const int*)  d_in[2];
    const int*   bat = (const int*)  d_in[3];
    const float* w1    = (const float*)d_in[4];
    const float* asrc1 = (const float*)d_in[5];
    const float* adst1 = (const float*)d_in[6];
    const float* b1    = (const float*)d_in[7];
    const float* elw1  = (const float*)d_in[8];
    const float* elb1  = (const float*)d_in[9];
    const float* pw1   = (const float*)d_in[10];
    const float* pb1   = (const float*)d_in[11];
    const float* bng1  = (const float*)d_in[12];
    const float* bnb1  = (const float*)d_in[13];
    const float* w2    = (const float*)d_in[14];
    const float* asrc2 = (const float*)d_in[15];
    const float* adst2 = (const float*)d_in[16];
    const float* b2    = (const float*)d_in[17];
    const float* elw2  = (const float*)d_in[18];
    const float* elb2  = (const float*)d_in[19];
    const float* pw2   = (const float*)d_in[20];
    const float* pb2   = (const float*)d_in[21];
    const float* bng2  = (const float*)d_in[22];
    const float* bnb2  = (const float*)d_in[23];
    const float* fw    = (const float*)d_in[24];
    const float* fb    = (const float*)d_in[25];
    float* out = (float*)d_out;

    const int N  = in_sizes[3];          // 6400
    const int E  = in_sizes[1];          // 1280000
    const int D1 = in_sizes[0] / N;      // 200
    const int NG = 32, NC = out_size / NG;

    // ---- workspace carving (256B aligned) ----
    char* ws = (char*)d_ws;
    size_t off = 0;
    auto take = [&](size_t bytes) {
        size_t p = off; off += (bytes + 255) & ~(size_t)255; return p;
    };
    float*    XP   = (float*)(ws + take((size_t)N * HID * 4));
    float*    Z1   = (float*)(ws + take((size_t)N * HID * 4));
    float*    Z2   = (float*)(ws + take((size_t)N * HID * 4));
    float*    AGG  = (float*)(ws + take((size_t)N * HID * 4));
    float*    ALS  = (float*)(ws + take((size_t)N * 4));
    float*    ALD  = (float*)(ws + take((size_t)N * 4));
    unsigned* MENC = (unsigned*)(ws + take((size_t)N * 4));
    float*    MF   = (float*)(ws + take((size_t)N * 4));
    float*    SSUM = (float*)(ws + take((size_t)N * 4));
    float*    ABUF = (float*)(ws + take((size_t)E * 4));
    float*    POOL = (float*)(ws + take((size_t)(NG * HID + NG) * 4));
    float*    CNT  = POOL + NG * HID;

    const int gN    = (N + 255) / 256;
    const int gNH   = (N * HID + 255) / 256;
    const int gE    = (E + 255) / 256;
    const int nTile = (N + 15) / 16;                         // 400
    const int gProj = (nTile + 3) / 4;                       // 4 waves/block
    const int eWav  = (E / 16 + MSG_TILES - 1) / MSG_TILES;  // 5000 waves
    const int gMsg  = (eWav + 7) / 8;                        // 8 waves/block

    // ================= block 1 =================
    clear_block_state<<<gNH, 256, 0, stream>>>(MENC, SSUM, AGG, N);
    if (D1 == 200)
        proj_wmma<200><<<gProj, 128, 0, stream>>>(x, w1, XP, N);
    else
        proj_wmma<HID><<<gProj, 128, 0, stream>>>(x, w1, XP, N);
    alpha_kernel<<<gN, 256, 0, stream>>>(XP, asrc1, adst1, ALS, ALD, N);
    edge_logits<<<gE, 256, 0, stream>>>(ei, ALS, ALD, ABUF, MENC, E);
    decode_m<<<gN, 256, 0, stream>>>(MENC, MF, N);
    edge_expsum<<<gE, 256, 0, stream>>>(ei, ABUF, MF, SSUM, E);
    msg_wmma<<<gMsg, 256, 0, stream>>>(ei, ea, ABUF, MF, SSUM, XP, elw1, elb1, AGG, E);
    tail_wmma<<<gProj, 128, 0, stream>>>(AGG, b1, pw1, pb1, bng1, bnb1, Z1, N);

    // ================= block 2 =================
    clear_block_state<<<gNH, 256, 0, stream>>>(MENC, SSUM, AGG, N);
    proj_wmma<HID><<<gProj, 128, 0, stream>>>(Z1, w2, XP, N);
    alpha_kernel<<<gN, 256, 0, stream>>>(XP, asrc2, adst2, ALS, ALD, N);
    edge_logits<<<gE, 256, 0, stream>>>(ei, ALS, ALD, ABUF, MENC, E);
    decode_m<<<gN, 256, 0, stream>>>(MENC, MF, N);
    edge_expsum<<<gE, 256, 0, stream>>>(ei, ABUF, MF, SSUM, E);
    msg_wmma<<<gMsg, 256, 0, stream>>>(ei, ea, ABUF, MF, SSUM, XP, elw2, elb2, AGG, E);
    tail_wmma<<<gProj, 128, 0, stream>>>(AGG, b2, pw2, pb2, bng2, bnb2, Z2, N);

    // ================= pool + head =================
    clear_pool<<<1, 256, 0, stream>>>(POOL, NG * HID + NG);
    pool_kernel<<<gNH, 256, 0, stream>>>(Z2, bat, POOL, CNT, N);
    head_kernel<<<1, 64, 0, stream>>>(POOL, CNT, fw, fb, out, NG, NC);
}
